// mutihead_self_attention_37220186587430
// MI455X (gfx1250) — matrix operand order repaired and checked by reading, not compiled, not run on hardware
//
#include <hip/hip_runtime.h>
#include <hip/hip_bf16.h>

#define D_MODEL 1024
#define SEQ     2048
#define BATCH   2
#define HEADS   16
#define DK      64
#define M_ROWS  (BATCH * SEQ)   // 4096

typedef __attribute__((ext_vector_type(16))) __bf16 v16bf;
typedef __attribute__((ext_vector_type(8)))  __bf16 v8bf;
typedef __attribute__((ext_vector_type(4)))  __bf16 v4bf;
typedef __attribute__((ext_vector_type(8)))  float  v8f;

union FragBF { v16bf v; v8bf h[2]; };

__device__ __forceinline__ v8f wmma_bf16(v16bf a, v16bf b, v8f c) {
  // D = A(16x32 bf16) * B(32x16 bf16) + C(16x16 f32)
  return __builtin_amdgcn_wmma_f32_16x16x32_bf16(false, a, false, b, (short)0, c,
                                                 false, false);
}

// ---------------------------------------------------------------------------
// Kernel 1: f32 -> bf16 convert, 4 elements per thread
// ---------------------------------------------------------------------------
__global__ void __launch_bounds__(256)
cvt_f32_bf16(const float* __restrict__ src, __bf16* __restrict__ dst, int n4) {
  int i = blockIdx.x * blockDim.x + threadIdx.x;
  if (i < n4) {
    const float4 v = ((const float4*)src)[i];
    v4bf o;
    o[0] = (__bf16)v.x; o[1] = (__bf16)v.y;
    o[2] = (__bf16)v.z; o[3] = (__bf16)v.w;
    ((v4bf*)dst)[i] = o;
  }
}

// ---------------------------------------------------------------------------
// Kernel 2: projection GEMM  Y[m,n] = sum_d X[m,d] * W[n,d]
//   vmode 0: store as [B,H,S,dk] bf16   (Q, K)
//   vmode 1: store as [B,H,dk,S] bf16   (V transposed for the P@V pass)
// One wave computes a 32x64 output tile; 8 WMMAs per K=32 step (12 b128 loads).
// ---------------------------------------------------------------------------
__global__ void __launch_bounds__(256)
proj_qkv_kernel(const __bf16* __restrict__ X, const __bf16* __restrict__ W,
                __bf16* __restrict__ out, int vmode) {
  const int lane = threadIdx.x & 31;
  const int wave = threadIdx.x >> 5;
  const int gw   = blockIdx.x * 8 + wave;          // 2048 waves total
  const int tM   = gw >> 4;                        // 0..127  (4096/32)
  const int tN   = gw & 15;                        // 0..15   (1024/64)
  const int m0   = tM * 32, n0 = tN * 64;
  const int lhi  = lane >> 4;                      // 0/1
  const int llo  = lane & 15;

  v8f acc[2][4] = {};

  const __bf16* arow[2];
  arow[0] = X + (size_t)(m0 + llo) * D_MODEL + lhi * 8;
  arow[1] = arow[0] + 16 * D_MODEL;
  const __bf16* brow[4];
#pragma unroll
  for (int ni = 0; ni < 4; ++ni)
    brow[ni] = W + (size_t)(n0 + ni * 16 + llo) * D_MODEL + lhi * 8;

  for (int k0 = 0; k0 < D_MODEL; k0 += 32) {
    FragBF a[2], b[4];
#pragma unroll
    for (int mi = 0; mi < 2; ++mi) {
      a[mi].h[0] = *(const v8bf*)(arow[mi] + k0);
      a[mi].h[1] = *(const v8bf*)(arow[mi] + k0 + 16);
    }
#pragma unroll
    for (int ni = 0; ni < 4; ++ni) {
      b[ni].h[0] = *(const v8bf*)(brow[ni] + k0);
      b[ni].h[1] = *(const v8bf*)(brow[ni] + k0 + 16);
    }
    __builtin_prefetch(arow[0] + k0 + 256, 0, 3);
    __builtin_prefetch(brow[0] + k0 + 256, 0, 3);
    __builtin_prefetch(brow[2] + k0 + 256, 0, 3);
#pragma unroll
    for (int mi = 0; mi < 2; ++mi)
#pragma unroll
      for (int ni = 0; ni < 4; ++ni)
        acc[mi][ni] = wmma_bf16(a[mi].v, b[ni].v, acc[mi][ni]);
  }

#pragma unroll
  for (int mi = 0; mi < 2; ++mi) {
#pragma unroll
    for (int ni = 0; ni < 4; ++ni) {
#pragma unroll
      for (int r = 0; r < 8; ++r) {
        const int m  = m0 + mi * 16 + r + lhi * 8;
        const int n  = n0 + ni * 16 + llo;
        const int bb = m >> 11, s = m & 2047;      // batch, seq
        const int h  = n >> 6,  d = n & 63;        // head, head-dim
        const __bf16 val = (__bf16)acc[mi][ni][r];
        if (vmode == 0)
          out[((size_t)(bb * HEADS + h) * SEQ + s) * DK + d] = val;
        else
          out[((size_t)(bb * HEADS + h) * DK + d) * SEQ + s] = val;
      }
    }
  }
}

// ---------------------------------------------------------------------------
// Kernel 3: causal flash attention. One wave per (b, h, 16-query tile).
// 32-key blocks: 4 WMMAs for the 16x32 score tile + 4 WMMAs for P@V.
// ---------------------------------------------------------------------------
__global__ void __launch_bounds__(256)
attn_kernel(const __bf16* __restrict__ Q, const __bf16* __restrict__ Km,
            const __bf16* __restrict__ Vt, __bf16* __restrict__ AO) {
  __shared__ __bf16 pbuf[8][16 * 32];              // per-wave P staging (8 KB)

  const int lane = threadIdx.x & 31;
  const int wave = threadIdx.x >> 5;
  const int gw   = blockIdx.x * 8 + wave;          // 4096 waves = B*H*(S/16)
  const int b    = gw >> 11;                       // / (16*128)
  const int h    = (gw >> 7) & 15;
  const int qt   = gw & 127;
  const int q0   = qt * 16;
  const int lhi  = lane >> 4;
  const int llo  = lane & 15;

  const __bf16* qhead = Q  + (size_t)(b * HEADS + h) * SEQ * DK;
  const __bf16* khead = Km + (size_t)(b * HEADS + h) * SEQ * DK;
  const __bf16* vhead = Vt + (size_t)(b * HEADS + h) * DK * SEQ;

  // Q fragments for dk=64 (two K=32 A-fragments), resident for the whole loop
  FragBF qa0, qa1;
  const __bf16* qrow = qhead + (size_t)(q0 + llo) * DK + lhi * 8;
  qa0.h[0] = *(const v8bf*)(qrow);
  qa0.h[1] = *(const v8bf*)(qrow + 16);
  qa1.h[0] = *(const v8bf*)(qrow + 32);
  qa1.h[1] = *(const v8bf*)(qrow + 48);

  v8f oacc[4] = {};
  float mrun[8], lrun[8];
#pragma unroll
  for (int r = 0; r < 8; ++r) { mrun[r] = -__builtin_inff(); lrun[r] = 0.f; }

  const float scale = 0.125f;                      // 1/sqrt(64)
  const float l2e   = 1.44269504f;

  const int nsteps = (q0 + 16 + 31) >> 5;
  for (int step = 0; step < nsteps; ++step) {
    const int kb0 = step * 32;

    // prefetch next key block (K rows and V rows) while this one computes
    if (step + 1 < nsteps) {
      __builtin_prefetch(khead + (size_t)(kb0 + 32 + lane) * DK, 0, 3);
      __builtin_prefetch(vhead + (size_t)(lane * 2) * SEQ + kb0 + 32, 0, 3);
    }

    // ---- scores S = Q K^T for a 16x32 key block (two 16x16 C fragments)
    v8f sc[2];
#pragma unroll
    for (int j = 0; j < 2; ++j) {
      const __bf16* krow = khead + (size_t)(kb0 + j * 16 + llo) * DK + lhi * 8;
      FragBF kf0, kf1;
      kf0.h[0] = *(const v8bf*)(krow);
      kf0.h[1] = *(const v8bf*)(krow + 16);
      kf1.h[0] = *(const v8bf*)(krow + 32);
      kf1.h[1] = *(const v8bf*)(krow + 48);
      v8f z = {0.f, 0.f, 0.f, 0.f, 0.f, 0.f, 0.f, 0.f};
      z = wmma_bf16(qa0.v, kf0.v, z);
      sc[j] = wmma_bf16(qa1.v, kf1.v, z);
    }

    // ---- scale + causal mask
#pragma unroll
    for (int j = 0; j < 2; ++j) {
#pragma unroll
      for (int r = 0; r < 8; ++r) {
        const int qi = q0 + r + lhi * 8;
        const int ki = kb0 + j * 16 + llo;
        const float s = sc[j][r] * scale;
        sc[j][r] = (ki <= qi) ? s : -__builtin_inff();
      }
    }

    // ---- online softmax (row stats reduced across the 16-lane half-group)
#pragma unroll
    for (int r = 0; r < 8; ++r) {
      float mx = fmaxf(sc[0][r], sc[1][r]);
      mx = fmaxf(mx, __shfl_xor(mx, 1, 32));
      mx = fmaxf(mx, __shfl_xor(mx, 2, 32));
      mx = fmaxf(mx, __shfl_xor(mx, 4, 32));
      mx = fmaxf(mx, __shfl_xor(mx, 8, 32));
      const float mnew = fmaxf(mrun[r], mx);
      const float corr = __builtin_exp2f((mrun[r] - mnew) * l2e);
      const float p0 = __builtin_exp2f((sc[0][r] - mnew) * l2e);
      const float p1 = __builtin_exp2f((sc[1][r] - mnew) * l2e);
      sc[0][r] = p0; sc[1][r] = p1;
      float rs = p0 + p1;
      rs += __shfl_xor(rs, 1, 32);
      rs += __shfl_xor(rs, 2, 32);
      rs += __shfl_xor(rs, 4, 32);
      rs += __shfl_xor(rs, 8, 32);
      lrun[r] = lrun[r] * corr + rs;
      mrun[r] = mnew;
#pragma unroll
      for (int t = 0; t < 4; ++t) oacc[t][r] *= corr;
    }

    // ---- stage P (C layout) into LDS, reload in A-fragment layout
    __bf16* pw = pbuf[wave];
#pragma unroll
    for (int j = 0; j < 2; ++j)
#pragma unroll
      for (int r = 0; r < 8; ++r)
        pw[(r + lhi * 8) * 32 + j * 16 + llo] = (__bf16)sc[j][r];

    asm volatile("s_wait_dscnt 0" ::: "memory");   // wave-internal DS ordering

    FragBF pf;
    const __bf16* pr = pw + llo * 32 + lhi * 8;
    pf.h[0] = *(const v8bf*)(pr);
    pf.h[1] = *(const v8bf*)(pr + 16);

    // ---- O += P @ V   (V stored transposed [dk, S] -> contiguous B frags)
#pragma unroll
    for (int t = 0; t < 4; ++t) {
      const __bf16* vrow = vhead + (size_t)(t * 16 + llo) * SEQ + kb0 + lhi * 8;
      FragBF vf;
      vf.h[0] = *(const v8bf*)(vrow);
      vf.h[1] = *(const v8bf*)(vrow + 16);
      oacc[t] = wmma_bf16(pf.v, vf.v, oacc[t]);
    }
  }

  // ---- normalize and store attention output as [B,S,D] bf16
#pragma unroll
  for (int t = 0; t < 4; ++t) {
#pragma unroll
    for (int r = 0; r < 8; ++r) {
      const int s    = q0 + r + lhi * 8;
      const int dcol = h * 64 + t * 16 + llo;
      const float o  = oacc[t][r] / lrun[r];
      AO[(size_t)(b * SEQ + s) * D_MODEL + dcol] = (__bf16)o;
    }
  }
}

// ---------------------------------------------------------------------------
// Kernel 4: output projection  out[m,n] = sum_d AO[m,d] * Wo[n,d]  (f32 out)
// 32x64 wave tile, same structure as proj_qkv_kernel.
// ---------------------------------------------------------------------------
__global__ void __launch_bounds__(256)
proj_out_kernel(const __bf16* __restrict__ AO, const __bf16* __restrict__ W,
                float* __restrict__ out) {
  const int lane = threadIdx.x & 31;
  const int wave = threadIdx.x >> 5;
  const int gw   = blockIdx.x * 8 + wave;
  const int tM   = gw >> 4;
  const int tN   = gw & 15;
  const int m0   = tM * 32, n0 = tN * 64;
  const int lhi  = lane >> 4;
  const int llo  = lane & 15;

  v8f acc[2][4] = {};

  const __bf16* arow[2];
  arow[0] = AO + (size_t)(m0 + llo) * D_MODEL + lhi * 8;
  arow[1] = arow[0] + 16 * D_MODEL;
  const __bf16* brow[4];
#pragma unroll
  for (int ni = 0; ni < 4; ++ni)
    brow[ni] = W + (size_t)(n0 + ni * 16 + llo) * D_MODEL + lhi * 8;

  for (int k0 = 0; k0 < D_MODEL; k0 += 32) {
    FragBF a[2], b[4];
#pragma unroll
    for (int mi = 0; mi < 2; ++mi) {
      a[mi].h[0] = *(const v8bf*)(arow[mi] + k0);
      a[mi].h[1] = *(const v8bf*)(arow[mi] + k0 + 16);
    }
#pragma unroll
    for (int ni = 0; ni < 4; ++ni) {
      b[ni].h[0] = *(const v8bf*)(brow[ni] + k0);
      b[ni].h[1] = *(const v8bf*)(brow[ni] + k0 + 16);
    }
    __builtin_prefetch(arow[0] + k0 + 256, 0, 3);
    __builtin_prefetch(brow[0] + k0 + 256, 0, 3);
    __builtin_prefetch(brow[2] + k0 + 256, 0, 3);
#pragma unroll
    for (int mi = 0; mi < 2; ++mi)
#pragma unroll
      for (int ni = 0; ni < 4; ++ni)
        acc[mi][ni] = wmma_bf16(a[mi].v, b[ni].v, acc[mi][ni]);
  }

#pragma unroll
  for (int mi = 0; mi < 2; ++mi)
#pragma unroll
    for (int ni = 0; ni < 4; ++ni)
#pragma unroll
      for (int r = 0; r < 8; ++r) {
        const int m = m0 + mi * 16 + r + lhi * 8;
        const int n = n0 + ni * 16 + llo;
        out[(size_t)m * D_MODEL + n] = acc[mi][ni][r];
      }
}

// ---------------------------------------------------------------------------
extern "C" void kernel_launch(void* const* d_in, const int* in_sizes, int n_in,
                              void* d_out, int out_size, void* d_ws, size_t ws_size,
                              hipStream_t stream) {
  const float* X  = (const float*)d_in[0];
  const float* Wq = (const float*)d_in[1];
  const float* Wk = (const float*)d_in[2];
  const float* Wv = (const float*)d_in[3];
  const float* Wo = (const float*)d_in[4];

  const size_t SZ_X = (size_t)M_ROWS * D_MODEL;                // 4M elems
  const size_t SZ_W = (size_t)D_MODEL * D_MODEL;               // 1M elems
  const size_t SZ_P = (size_t)BATCH * HEADS * SEQ * DK;        // 4M elems

  __bf16* Xb  = (__bf16*)d_ws;
  __bf16* Wqb = Xb  + SZ_X;
  __bf16* Wkb = Wqb + SZ_W;
  __bf16* Wvb = Wkb + SZ_W;
  __bf16* Wob = Wvb + SZ_W;
  __bf16* Qb  = Wob + SZ_W;
  __bf16* Kb  = Qb  + SZ_P;
  __bf16* Vtb = Kb  + SZ_P;
  __bf16* AOb = Vtb + SZ_P;                                    // total ~48 MB

  // 1) f32 -> bf16 converts
  cvt_f32_bf16<<<(int)(SZ_X / 4 / 256), 256, 0, stream>>>(X,  Xb,  (int)(SZ_X / 4));
  cvt_f32_bf16<<<(int)(SZ_W / 4 / 256), 256, 0, stream>>>(Wq, Wqb, (int)(SZ_W / 4));
  cvt_f32_bf16<<<(int)(SZ_W / 4 / 256), 256, 0, stream>>>(Wk, Wkb, (int)(SZ_W / 4));
  cvt_f32_bf16<<<(int)(SZ_W / 4 / 256), 256, 0, stream>>>(Wv, Wvb, (int)(SZ_W / 4));
  cvt_f32_bf16<<<(int)(SZ_W / 4 / 256), 256, 0, stream>>>(Wo, Wob, (int)(SZ_W / 4));

  // 2) Q, K (as [B,H,S,dk]) and V (transposed [B,H,dk,S]) projections
  proj_qkv_kernel<<<256, 256, 0, stream>>>(Xb, Wqb, Qb,  0);
  proj_qkv_kernel<<<256, 256, 0, stream>>>(Xb, Wkb, Kb,  0);
  proj_qkv_kernel<<<256, 256, 0, stream>>>(Xb, Wvb, Vtb, 1);

  // 3) causal flash attention -> AO [B,S,D] bf16
  attn_kernel<<<512, 256, 0, stream>>>(Qb, Kb, Vtb, AOb);

  // 4) output projection -> f32 result
  proj_out_kernel<<<256, 256, 0, stream>>>(AOb, Wob, (float*)d_out);
}
